// SquatContext_44014824849811
// MI455X (gfx1250) — matrix operand order repaired and checked by reading, not compiled, not run on hardware
//
#include <hip/hip_runtime.h>

// ---------------------------------------------------------------------------
// Problem dimensions (fixed by the reference)
// ---------------------------------------------------------------------------
#define DMODEL 1024
#define NHEAD  8
#define HDIM   128
#define FFDIM  1024
#define ES_N   2048
#define NN_N   1024
#define EP_N   16384
#define KG_N   4096

typedef __attribute__((ext_vector_type(16))) __bf16 v16bf;
typedef __attribute__((ext_vector_type(8)))  float  v8f;

// ---------------------------------------------------------------------------
// bf16 helpers (round-to-nearest-even f32 -> bf16)
// ---------------------------------------------------------------------------
__device__ __forceinline__ unsigned short f2bf_bits(float f) {
  unsigned int u = __builtin_bit_cast(unsigned int, f);
  u += 0x7fffu + ((u >> 16) & 1u);
  return (unsigned short)(u >> 16);
}
__device__ __forceinline__ __bf16 bits2bf(unsigned short s) {
  return __builtin_bit_cast(__bf16, s);
}
__device__ __forceinline__ v8f wmma_bf16(v16bf a, v16bf b, v8f c) {
  // D(16x16 f32) = A(16x32 bf16) * B(32x16 bf16) + C
  return __builtin_amdgcn_wmma_f32_16x16x32_bf16(false, a, false, b,
                                                 (short)0, c, false, false);
}
__device__ __forceinline__ v8f zero8() {
  v8f z;
#pragma unroll
  for (int i = 0; i < 8; ++i) z[i] = 0.f;
  return z;
}

// ---------------------------------------------------------------------------
// WMMA operand gathers from LDS (ushort = bf16 bits)
// A 16x32 (MxK): lane m = lane&15; elem e -> k = e + 16*hi
// ---------------------------------------------------------------------------
__device__ __forceinline__ v16bf gather_A(const unsigned short* s, int ld, int lane) {
  int m = lane & 15, hi = (lane >> 4) & 1;
  const unsigned short* r = s + m * ld + hi * 16;
  v16bf a;
#pragma unroll
  for (int e = 0; e < 16; ++e) a[e] = bits2bf(r[e]);
  return a;
}
// B 32x16 (KxN), source stored [n][k] row-major (i.e. B = srcT):
// lane n = lane&15; elem e -> k = e + 16*hi
__device__ __forceinline__ v16bf gather_B_nk(const unsigned short* s, int ld, int lane) {
  int n = lane & 15, hi = (lane >> 4) & 1;
  const unsigned short* r = s + n * ld + hi * 16;
  v16bf b;
#pragma unroll
  for (int e = 0; e < 16; ++e) b[e] = bits2bf(r[e]);
  return b;
}
// B 32x16 (KxN), source stored [k][n] row-major:
__device__ __forceinline__ v16bf gather_B_kn(const unsigned short* s, int ld, int lane) {
  int n = lane & 15, hi = (lane >> 4) & 1;
  const unsigned short* r = s + hi * 16 * ld + n;
  v16bf b;
#pragma unroll
  for (int e = 0; e < 16; ++e) b[e] = bits2bf(r[e * ld]);
  return b;
}

// ---------------------------------------------------------------------------
// GEMM: Y[M,N] = X[M,K] @ W[N,K]^T (+bias) (+res) (relu)
// 256 threads = 8 waves; block tile 128x128; wave tile 32x64 (2x4 WMMA tiles).
// Software-pipelined: global loads for chunk k+1 issue before the WMMAs of
// chunk k; ping-pong LDS buffers give one barrier per k-chunk. f32 -> bf16
// conversion happens in the register->LDS stage, under the WMMA shadow.
// ---------------------------------------------------------------------------
#define GBM 128
#define GBK 32
#define GLD 34   // padded ushort row stride

template <bool HAS_BIAS, bool HAS_RES, bool RELU>
__global__ __launch_bounds__(256) void gemm_kernel(
    const float* __restrict__ X, const float* __restrict__ W,
    const float* __restrict__ bias, const float* __restrict__ res,
    float* __restrict__ Y, int M, int N, int K) {
  __shared__ unsigned short As[2][GBM * GLD];
  __shared__ unsigned short Bs[2][GBM * GLD];
  const int tid  = threadIdx.x;
  const int lane = tid & 31;
  const int wid  = tid >> 5;
  const int wm   = wid >> 1;       // 0..3
  const int wn   = wid & 1;        // 0..1
  const int m0   = blockIdx.y * GBM;
  const int n0   = blockIdx.x * GBM;
  const int srow = tid >> 3;              // staging row   (0..31 per 1024-elem slab)
  const int scol = (tid & 7) * 4;         // staging col*4 (vectorizable 4-float runs)

  float ra[16], rb[16];
  auto ldg = [&](int kb) {
#pragma unroll
    for (int i = 0; i < 4; ++i) {         // 4 row-groups of 32 rows
#pragma unroll
      for (int v = 0; v < 4; ++v) {
        ra[i * 4 + v] = X[(size_t)(m0 + i * 32 + srow) * K + kb + scol + v];
        rb[i * 4 + v] = W[(size_t)(n0 + i * 32 + srow) * K + kb + scol + v];
      }
    }
  };
  auto sts = [&](int b) {
#pragma unroll
    for (int i = 0; i < 4; ++i) {
#pragma unroll
      for (int v = 0; v < 4; ++v) {
        As[b][(i * 32 + srow) * GLD + scol + v] = f2bf_bits(ra[i * 4 + v]);
        Bs[b][(i * 32 + srow) * GLD + scol + v] = f2bf_bits(rb[i * 4 + v]);
      }
    }
  };

  v8f acc[2][4];
#pragma unroll
  for (int i = 0; i < 2; ++i)
#pragma unroll
    for (int j = 0; j < 4; ++j) acc[i][j] = zero8();

  ldg(0);
  sts(0);
  int buf = 0;
  for (int kb = 0; kb < K; kb += GBK) {
    __syncthreads();                       // LDS[buf] ready for all waves
    const bool more = (kb + GBK) < K;
    if (more) ldg(kb + GBK);               // hide HBM/L2 latency under WMMAs
    v16bf af[2], bfr[4];
#pragma unroll
    for (int i = 0; i < 2; ++i)
      af[i] = gather_A(&As[buf][(wm * 32 + i * 16) * GLD], GLD, lane);
#pragma unroll
    for (int j = 0; j < 4; ++j)
      bfr[j] = gather_B_nk(&Bs[buf][(wn * 64 + j * 16) * GLD], GLD, lane);
#pragma unroll
    for (int i = 0; i < 2; ++i)
#pragma unroll
      for (int j = 0; j < 4; ++j)
        acc[i][j] = wmma_bf16(af[i], bfr[j], acc[i][j]);
    if (more) sts(buf ^ 1);                // disjoint buffer: no barrier needed
    buf ^= 1;
  }

  const int nl = lane & 15, hi = (lane >> 4) & 1;
#pragma unroll
  for (int i = 0; i < 2; ++i) {
#pragma unroll
    for (int j = 0; j < 4; ++j) {
      int n = n0 + wn * 64 + j * 16 + nl;
      float bv = HAS_BIAS ? bias[n] : 0.f;
#pragma unroll
      for (int r = 0; r < 8; ++r) {
        int m = m0 + wm * 32 + i * 16 + r + 8 * hi;
        float v = acc[i][j][r] + bv;
        if constexpr (HAS_RES) v += res[(size_t)m * N + n];
        if constexpr (RELU)    v = v > 0.f ? v : 0.f;
        Y[(size_t)m * N + n] = v;
      }
    }
  }
}

// ---------------------------------------------------------------------------
// Flash attention: grid (Lq/64, H); 128 threads = 4 waves; wave owns 16 q rows
// of one head. K/V staged 32 rows at a time into LDS as bf16; next chunk is
// prefetched (global_prefetch_b8) while the current chunk runs softmax+WMMA.
// ---------------------------------------------------------------------------
#define FLD 132  // padded ushort stride for 128-wide K/V chunks

__global__ __launch_bounds__(128) void flash_attn_kernel(
    const float* __restrict__ Qp, const float* __restrict__ Kp,
    const float* __restrict__ Vp, float* __restrict__ Out, int Lkv) {
  __shared__ unsigned short Kc[32 * FLD];
  __shared__ unsigned short Vc[32 * FLD];
  __shared__ unsigned short Pst[4][16 * GLD];

  const int tid = threadIdx.x, lane = tid & 31, w = tid >> 5;
  const int h = blockIdx.y;
  const int q0 = blockIdx.x * 64 + w * 16;
  const int col0 = h * HDIM;
  const int nl = lane & 15, hi = (lane >> 4) & 1;
  const float scl = 0.08838834764831845f;  // 1/sqrt(128)

  // Resident Q A-frags: 16 rows x 128 cols = 4 frags of 16x32
  v16bf qf[4];
#pragma unroll
  for (int kc = 0; kc < 4; ++kc) {
    const float* src = Qp + (size_t)(q0 + nl) * DMODEL + col0 + kc * 32 + hi * 16;
    v16bf a;
#pragma unroll
    for (int e = 0; e < 16; ++e)
      a[e] = bits2bf(f2bf_bits(src[e]));
    qf[kc] = a;
  }

  v8f o[8];
#pragma unroll
  for (int t = 0; t < 8; ++t) o[t] = zero8();
  float rmax[8], rsum[8];
#pragma unroll
  for (int r = 0; r < 8; ++r) { rmax[r] = -1e30f; rsum[r] = 0.f; }

  for (int c0 = 0; c0 < Lkv; c0 += 32) {
#pragma unroll
    for (int i = 0; i < 32; ++i) {               // 4096 elems / 128 thr
      int idx = i * 128 + tid;
      int r = idx >> 7, cc = idx & 127;
      Kc[r * FLD + cc] = f2bf_bits(Kp[(size_t)(c0 + r) * DMODEL + col0 + cc]);
      Vc[r * FLD + cc] = f2bf_bits(Vp[(size_t)(c0 + r) * DMODEL + col0 + cc]);
    }
    if (c0 + 32 < Lkv) {
      // 128 threads x one 128B line each == the full 16KB next chunk (x2)
      size_t poff = (size_t)(c0 + 32 + (tid >> 2)) * DMODEL + col0 + (tid & 3) * 32;
      __builtin_prefetch(Kp + poff, 0, 3);
      __builtin_prefetch(Vp + poff, 0, 3);
    }
    __syncthreads();

    // S(16x32) = Q(16x128) @ Kchunk^T : two 16x16 C tiles, 4 K-steps each
    v8f s0 = zero8(), s1 = zero8();
#pragma unroll
    for (int kc = 0; kc < 4; ++kc) {
      s0 = wmma_bf16(qf[kc], gather_B_nk(&Kc[0 * FLD + kc * 32], FLD, lane), s0);
      s1 = wmma_bf16(qf[kc], gather_B_nk(&Kc[16 * FLD + kc * 32], FLD, lane), s1);
    }

    // Online softmax (row m = r + 8*hi; reduce across 16 lanes of the half)
    float alpha[8];
#pragma unroll
    for (int r = 0; r < 8; ++r) {
      float a0 = s0[r] * scl, a1 = s1[r] * scl;
      float mx = fmaxf(a0, a1);
#pragma unroll
      for (int d = 1; d < 16; d <<= 1) mx = fmaxf(mx, __shfl_xor(mx, d, 32));
      float nm = fmaxf(rmax[r], mx);
      alpha[r] = __expf(rmax[r] - nm);
      rmax[r] = nm;
      float p0 = __expf(a0 - nm), p1 = __expf(a1 - nm);
      float ps = p0 + p1;
#pragma unroll
      for (int d = 1; d < 16; d <<= 1) ps += __shfl_xor(ps, d, 32);
      rsum[r] = rsum[r] * alpha[r] + ps;
      s0[r] = p0; s1[r] = p1;
    }
#pragma unroll
    for (int t = 0; t < 8; ++t)
#pragma unroll
      for (int r = 0; r < 8; ++r) o[t][r] *= alpha[r];

    // Restage P (C layout -> A layout) via per-wave LDS scratch
#pragma unroll
    for (int r = 0; r < 8; ++r) {
      int m = r + 8 * hi;
      Pst[w][m * GLD + nl]      = f2bf_bits(s0[r]);
      Pst[w][m * GLD + 16 + nl] = f2bf_bits(s1[r]);
    }
    asm volatile("s_wait_dscnt 0" ::: "memory");
    v16bf pf = gather_A(&Pst[w][0], GLD, lane);

    // O(16x128) += P(16x32) @ Vchunk(32x128)
#pragma unroll
    for (int t = 0; t < 8; ++t)
      o[t] = wmma_bf16(pf, gather_B_kn(&Vc[t * 16], FLD, lane), o[t]);
    __syncthreads();
  }

#pragma unroll
  for (int r = 0; r < 8; ++r) {
    float inv = 1.0f / rsum[r];
    int m = q0 + r + 8 * hi;
#pragma unroll
    for (int t = 0; t < 8; ++t)
      Out[(size_t)m * DMODEL + col0 + t * 16 + nl] = o[t][r] * inv;
  }
}

// ---------------------------------------------------------------------------
// LayerNorm: one block per row, y = (x-mu)*rstd*g + b
// ---------------------------------------------------------------------------
__global__ __launch_bounds__(256) void ln_kernel(
    const float* __restrict__ x, const float* __restrict__ g,
    const float* __restrict__ b, float* __restrict__ y) {
  const int row = blockIdx.x, tid = threadIdx.x;
  const float* xr = x + (size_t)row * DMODEL;
  float v[4], s = 0.f, ss = 0.f;
#pragma unroll
  for (int i = 0; i < 4; ++i) {
    v[i] = xr[tid + i * 256];
    s += v[i]; ss += v[i] * v[i];
  }
#pragma unroll
  for (int d = 1; d < 32; d <<= 1) {
    s += __shfl_xor(s, d, 32);
    ss += __shfl_xor(ss, d, 32);
  }
  __shared__ float shs[8], shss[8];
  if ((tid & 31) == 0) { shs[tid >> 5] = s; shss[tid >> 5] = ss; }
  __syncthreads();
  s = 0.f; ss = 0.f;
#pragma unroll
  for (int i = 0; i < 8; ++i) { s += shs[i]; ss += shss[i]; }
  float mu = s * (1.0f / DMODEL);
  float var = ss * (1.0f / DMODEL) - mu * mu;
  float rstd = rsqrtf(var + 1e-5f);
#pragma unroll
  for (int i = 0; i < 4; ++i) {
    int c = tid + i * 256;
    y[(size_t)row * DMODEL + c] = (v[i] - mu) * rstd * g[c] + b[c];
  }
}

// ---------------------------------------------------------------------------
// Index helpers: invmap[EP] (-1 or position in ind_pair), then row gather
// pair[ind[k]] without materializing the 64MB pair tensor.
// ---------------------------------------------------------------------------
__global__ void fill_neg1_kernel(int* __restrict__ p, int n) {
  int i = blockIdx.x * blockDim.x + threadIdx.x;
  if (i < n) p[i] = -1;
}
__global__ void scatter_inv_kernel(const int* __restrict__ ind, int* __restrict__ inv, int n) {
  int i = blockIdx.x * blockDim.x + threadIdx.x;
  if (i < n) inv[ind[i]] = i;
}
__global__ __launch_bounds__(256) void gather_pair_kernel(
    const int* __restrict__ ind, const int* __restrict__ inv,
    const float* __restrict__ sp, const float* __restrict__ entire,
    float* __restrict__ out) {
  int k = blockIdx.x;
  int row = ind[k];
  int p = inv[row];
  const float* src = (p >= 0) ? (sp + (size_t)p * DMODEL)
                              : (entire + (size_t)row * DMODEL);
  for (int i = threadIdx.x; i < DMODEL; i += 256)
    out[(size_t)k * DMODEL + i] = src[i];
}

// ---------------------------------------------------------------------------
// Host-side orchestration
// ---------------------------------------------------------------------------
static inline void launch_gemm(hipStream_t s, const float* X, const float* W,
                               const float* bias, const float* res, float* Y,
                               int M, int N, int K, int relu) {
  dim3 g(N / 128, M / 128), b(256);
  if (res)
    gemm_kernel<true, true, false><<<g, b, 0, s>>>(X, W, bias, res, Y, M, N, K);
  else if (relu)
    gemm_kernel<true, false, true><<<g, b, 0, s>>>(X, W, bias, res, Y, M, N, K);
  else
    gemm_kernel<true, false, false><<<g, b, 0, s>>>(X, W, bias, res, Y, M, N, K);
}

static inline void run_mha(hipStream_t s, const float* q, int Lq, const float* kv, int Lkv,
                           const float* Win, const float* bin,
                           const float* Wout, const float* bout,
                           const float* res, float* out,
                           float* qb, float* kb, float* vb, float* ao) {
  launch_gemm(s, q,  Win,                             bin,            nullptr, qb, Lq,  DMODEL, DMODEL, 0);
  launch_gemm(s, kv, Win + (size_t)DMODEL * DMODEL,   bin + DMODEL,   nullptr, kb, Lkv, DMODEL, DMODEL, 0);
  launch_gemm(s, kv, Win + (size_t)2 * DMODEL * DMODEL, bin + 2 * DMODEL, nullptr, vb, Lkv, DMODEL, DMODEL, 0);
  flash_attn_kernel<<<dim3(Lq / 64, NHEAD), dim3(128), 0, s>>>(qb, kb, vb, ao, Lkv);
  launch_gemm(s, ao, Wout, bout, res, out, Lq, DMODEL, DMODEL, 0);
}

extern "C" void kernel_launch(void* const* d_in, const int* in_sizes, int n_in,
                              void* d_out, int out_size, void* d_ws, size_t ws_size,
                              hipStream_t stream) {
  (void)in_sizes; (void)n_in; (void)out_size; (void)ws_size;
  const float* in_sp     = (const float*)d_in[0];
  const float* in_su     = (const float*)d_in[1];
  const float* entire    = (const float*)d_in[2];
  const int*   ind_pair  = (const int*)d_in[3];
  const int*   ind_e2e   = (const int*)d_in[4];
  const int*   ind_n2e   = (const int*)d_in[5];
  const float* Win       = (const float*)d_in[6];
  const float* bin       = (const float*)d_in[7];
  const float* Wout      = (const float*)d_in[8];
  const float* bout      = (const float*)d_in[9];
  const float* feW1      = (const float*)d_in[10];
  const float* feb1      = (const float*)d_in[11];
  const float* feW2      = (const float*)d_in[12];
  const float* feb2      = (const float*)d_in[13];
  const float* fnW1      = (const float*)d_in[14];
  const float* fnb1      = (const float*)d_in[15];
  const float* fnW2      = (const float*)d_in[16];
  const float* fnb2      = (const float*)d_in[17];
  const float* lng       = (const float*)d_in[18];
  const float* lnb       = (const float*)d_in[19];

  float* out_uu = (float*)d_out;                       // [NN, D]
  float* out_up = out_uu + (size_t)NN_N * DMODEL;      // [ES, D]

  // Workspace layout (~116 MB of f32 + 64 KB int)
  float* f = (float*)d_ws;
  size_t off = 0;
  auto alloc = [&](size_t n) { float* p = f + off; off += n; return p; };
  float* sp   = alloc((size_t)ES_N * DMODEL);
  float* su   = alloc((size_t)NN_N * DMODEL);
  float* qb   = alloc((size_t)ES_N * DMODEL);
  float* kb   = alloc((size_t)KG_N * DMODEL);
  float* vb   = alloc((size_t)KG_N * DMODEL);
  float* ao   = alloc((size_t)ES_N * DMODEL);
  float* t1   = alloc((size_t)ES_N * DMODEL);
  float* t2   = alloc((size_t)ES_N * DMODEL);
  float* e2e  = alloc((size_t)KG_N * DMODEL);
  float* n2e  = alloc((size_t)KG_N * DMODEL);
  float* hbuf = alloc((size_t)ES_N * FFDIM);
  int*   inv  = (int*)(f + off);

  auto WinI  = [&](int i) { return Win  + (size_t)i * 3 * DMODEL * DMODEL; };
  auto binI  = [&](int i) { return bin  + (size_t)i * 3 * DMODEL; };
  auto WoutI = [&](int i) { return Wout + (size_t)i * DMODEL * DMODEL; };
  auto boutI = [&](int i) { return bout + (size_t)i * DMODEL; };
  auto lnG   = [&](int i) { return lng + (size_t)i * DMODEL; };
  auto lnB   = [&](int i) { return lnb + (size_t)i * DMODEL; };

  // --- sp = LN(x + mha0(x,x)) ; su = LN(u + mha1(u,u)) ----------------------
  run_mha(stream, in_sp, ES_N, in_sp, ES_N, WinI(0), binI(0), WoutI(0), boutI(0), in_sp, t1, qb, kb, vb, ao);
  ln_kernel<<<ES_N, 256, 0, stream>>>(t1, lnG(0), lnB(0), sp);
  run_mha(stream, in_su, NN_N, in_su, NN_N, WinI(1), binI(1), WoutI(1), boutI(1), in_su, t1, qb, kb, vb, ao);
  ln_kernel<<<NN_N, 256, 0, stream>>>(t1, lnG(3), lnB(3), su);

  // --- scatter/gather: pair = entire.at[ind_pair].set(sp); gather e2e/n2e --
  fill_neg1_kernel<<<(EP_N + 255) / 256, 256, 0, stream>>>(inv, EP_N);
  scatter_inv_kernel<<<(ES_N + 255) / 256, 256, 0, stream>>>(ind_pair, inv, ES_N);
  gather_pair_kernel<<<KG_N, 256, 0, stream>>>(ind_e2e, inv, sp, entire, e2e);
  gather_pair_kernel<<<KG_N, 256, 0, stream>>>(ind_n2e, inv, sp, entire, n2e);

  // --- edge update: up ------------------------------------------------------
  run_mha(stream, sp, ES_N, e2e, KG_N, WinI(2), binI(2), WoutI(2), boutI(2), sp, t1, qb, kb, vb, ao);
  run_mha(stream, sp, ES_N, su,  NN_N, WinI(3), binI(3), WoutI(3), boutI(3), t1, t2, qb, kb, vb, ao);
  ln_kernel<<<ES_N, 256, 0, stream>>>(t2, lnG(1), lnB(1), t1);          // up0
  launch_gemm(stream, t1, feW1, feb1, nullptr, hbuf, ES_N, FFDIM, DMODEL, 1);
  launch_gemm(stream, hbuf, feW2, feb2, t1, t2, ES_N, DMODEL, FFDIM, 0);
  ln_kernel<<<ES_N, 256, 0, stream>>>(t2, lnG(2), lnB(2), out_up);

  // --- node update: uu ------------------------------------------------------
  run_mha(stream, su, NN_N, n2e, KG_N, WinI(4), binI(4), WoutI(4), boutI(4), su, t1, qb, kb, vb, ao);
  run_mha(stream, su, NN_N, su,  NN_N, WinI(5), binI(5), WoutI(5), boutI(5), t1, t2, qb, kb, vb, ao);
  ln_kernel<<<NN_N, 256, 0, stream>>>(t2, lnG(1), lnB(1), t1);          // uu0
  launch_gemm(stream, t1, fnW1, fnb1, nullptr, hbuf, NN_N, FFDIM, DMODEL, 1);
  launch_gemm(stream, hbuf, fnW2, fnb2, t1, t2, NN_N, DMODEL, FFDIM, 0);
  ln_kernel<<<NN_N, 256, 0, stream>>>(t2, lnG(2), lnB(2), out_uu);
}
